// CNN_70214125355555
// MI455X (gfx1250) — matrix operand order repaired and checked by reading, not compiled, not run on hardware
//
#include <hip/hip_runtime.h>
#include <math.h>
#include <stdint.h>

typedef __attribute__((ext_vector_type(16))) _Float16     v16h;
typedef __attribute__((ext_vector_type(8)))  _Float16     v8h;
typedef __attribute__((ext_vector_type(8)))  float        v8f;
typedef __attribute__((ext_vector_type(4)))  unsigned int v4u;
typedef __attribute__((ext_vector_type(8)))  int          v8i;
typedef __attribute__((ext_vector_type(4)))  int          v4i;

#define TPB 256
#define JAC_ITERS 100
#define LN_EPS 1e-5f

// ---------------- LDS layout (byte offsets into one blob) ----------------
#define OFF_R0    0          // field f32 (64KB) then out1 f16 [16*64*64] (128KB)
#define OFF_NORM  131072     // norm f16 [128*128]   (32768 B)
#define OFF_OUT2  163840     // out2 f16 [32*32*32]  (65536 B)
#define OFF_W1    229376     // w1 f16 [16][32]  zero-padded K  (1024 B)
#define OFF_W2    230400     // w2 f16 [32][160] zero-padded K  (10240 B)
#define OFF_W3    240640     // w3 f16 [64][288]                (36864 B)
#define OFF_ROWM  277504     // row_mean f32 [128]
#define OFF_COLM  278016     // col_mean f32 [128]
#define OFF_RED   278528     // reduce scratch f32 [32]
#define OFF_B1    278656     // b1 f32 [16]
#define OFF_B2    278720     // b2 f32 [32]
#define OFF_B3    278848     // b3 f32 [64]
#define OFF_FEAT  279104     // feat f32 [64]
#define OFF_OUT5  279360     // out5 f32 [8]
#define OFF_ZERO  279392     // zeroed slot for branchless OOB gathers
#define SM_TOTAL  279424

// half-element indices relative to LDS base
#define OUT1_H  0
#define NORM_H  (OFF_NORM / 2)
#define OUT2_H  (OFF_OUT2 / 2)
#define ZERO_H  (OFF_ZERO / 2)

// branchless tanh-GELU using native v_exp_f32 (exp2) and v_rcp_f32
__device__ __forceinline__ float gelu_tanh(float x) {
    const float c = 0.7978845608028654f;          // sqrt(2/pi)
    float t = c * (x + 0.044715f * x * x * x);
    t = fminf(fmaxf(t, -9.0f), 9.0f);
    float z  = __builtin_amdgcn_exp2f(t * 2.8853900817779268f);
    float th = 1.0f - 2.0f * __builtin_amdgcn_rcpf(z + 1.0f);
    return 0.5f * x * (1.0f + th);
}
__device__ __forceinline__ float nan_to_0(float x) {
    return (x != x || fabsf(x) > 3.402823466e38f) ? 0.0f : x;
}
__device__ __forceinline__ float nan_to_1(float x) {
    return (x != x || fabsf(x) > 3.402823466e38f) ? 1.0f : x;
}
// A-fragment K offset, contiguous order matching the B loads: K = h + 16*(lane>=16)
__device__ __forceinline__ int a_koff(int h, int lhi) {
    return h + (lhi << 4);
}

__global__ __launch_bounds__(TPB, 1) void encoder_fused_kernel(
    const float* __restrict__ sst,   // [B,128,128]
    const float* __restrict__ ln_w,  // [128,128]
    const float* __restrict__ ln_b,  // [128,128]
    const float* __restrict__ w1,    // [16,1,5,5]
    const float* __restrict__ b1,    // [16]
    const float* __restrict__ w2,    // [32,16,3,3]
    const float* __restrict__ b2,    // [32]
    const float* __restrict__ w3,    // [64,32,3,3]
    const float* __restrict__ b3,    // [64]
    const float* __restrict__ wl,    // [5,64]
    const float* __restrict__ bl,    // [5]
    float* __restrict__ out,         // mu [B,2] then sigma [B,2,2]
    int B)
{
    __shared__ __align__(16) char smem[SM_TOTAL];
    float*          field  = (float*)(smem + OFF_R0);
    _Float16*       out1h  = (_Float16*)(smem + OFF_R0);
    _Float16*       normh  = (_Float16*)(smem + OFF_NORM);
    _Float16*       out2h  = (_Float16*)(smem + OFF_OUT2);
    _Float16*       w1h    = (_Float16*)(smem + OFF_W1);
    _Float16*       w2h    = (_Float16*)(smem + OFF_W2);
    _Float16*       w3h    = (_Float16*)(smem + OFF_W3);
    const _Float16* lds_h  = (const _Float16*)smem;    // whole-LDS half view
    float*          row_m  = (float*)(smem + OFF_ROWM);
    float*          col_m  = (float*)(smem + OFF_COLM);
    float*          red    = (float*)(smem + OFF_RED);
    float*          b1s    = (float*)(smem + OFF_B1);
    float*          b2s    = (float*)(smem + OFF_B2);
    float*          b3s    = (float*)(smem + OFF_B3);
    float*          feat   = (float*)(smem + OFF_FEAT);
    float*          out5s  = (float*)(smem + OFF_OUT5);

    const int tid  = threadIdx.x;
    const int lane = tid & 31;
    const int wid  = tid >> 5;
    const int lhi  = lane >> 4;           // 0: lanes 0-15, 1: lanes 16-31
    const int b    = blockIdx.x;
    if (b >= B) return;

    // ---------- TDM: DMA the 128x128 f32 field tile into LDS (wave 0) -----
    if (wid == 0) {
        unsigned long long ga = (unsigned long long)(sst + (size_t)b * 16384);
        v4u g0;
        g0[0] = 1u;                                     // count=1 valid user D#
        g0[1] = (unsigned)OFF_R0;                       // lds_addr
        g0[2] = (unsigned)(ga & 0xFFFFFFFFu);           // global_addr[31:0]
        g0[3] = (unsigned)((ga >> 32) & 0x1FFFFFFu)     // global_addr[56:32]
              | (2u << 30);                             // type = 2 ("image")
        v8i g1;
        g1[0] = 0x00020000;                             // data_size = 4B
        g1[1] = (int)(128u << 16);                      // tensor_dim0 = 128
        g1[2] = (int)(128u << 16);                      // tensor_dim1 = 128
        g1[3] = (int)(128u << 16);                      // tile_dim0 = 128
        g1[4] = 128;                                    // tile_dim1=128, tile_dim2=0
        g1[5] = 128;                                    // tensor_dim0_stride = 128
        g1[6] = (int)(16384u << 16);                    // tensor_dim1_stride = 16384
        g1[7] = 0;
        v4i z4 = {0, 0, 0, 0};
#if defined(__clang_major__) && (__clang_major__ >= 23)
        v8i z8 = {0, 0, 0, 0, 0, 0, 0, 0};
        __builtin_amdgcn_tensor_load_to_lds(g0, g1, z4, z4, z8, 0);
#else
        __builtin_amdgcn_tensor_load_to_lds(g0, g1, z4, z4, 0);
#endif
        __builtin_amdgcn_s_wait_tensorcnt(0);
    }

    // ---------- stage weights / biases (f32 -> f16, K-padded) -------------
    // (runs on all waves concurrently with the TDM transfer)
    for (int i = tid; i < 16 * 32; i += TPB) {          // w1 -> [16][32]
        int n = i >> 5, k = i & 31;
        w1h[i] = (k < 25) ? (_Float16)w1[n * 25 + k] : (_Float16)0.0f;
    }
    for (int i = tid; i < 32 * 160; i += TPB) {         // w2 -> [32][160]
        int n = i / 160, k = i - n * 160;
        w2h[i] = (k < 144) ? (_Float16)w2[n * 144 + k] : (_Float16)0.0f;
    }
    for (int i = tid; i < 64 * 288; i += TPB)           // w3 -> [64][288]
        w3h[i] = (_Float16)w3[i];
    if (tid < 16) b1s[tid] = b1[tid];
    if (tid < 32) b2s[tid] = b2[tid];
    if (tid < 64) { b3s[tid] = b3[tid]; feat[tid] = 0.0f; }
    if (tid == 0) *(float*)(smem + OFF_ZERO) = 0.0f;
    __syncthreads();

    // ---------- row / column nanmeans ----------
    if (tid < 128) {
        float s = 0.0f; int cnt = 0;
        for (int c = 0; c < 128; ++c) {
            float v = field[tid * 128 + c];
            if (!(v != v)) { s += v; ++cnt; }
        }
        row_m[tid] = s / (float)cnt;
        s = 0.0f; cnt = 0;
        for (int r = 0; r < 128; ++r) {
            float v = field[r * 128 + tid];
            if (!(v != v)) { s += v; ++cnt; }
        }
        col_m[tid] = s / (float)cnt;
    }
    __syncthreads();

    // ---------- init fill + NaN mask; keep owned cells in registers -------
    // thread owns cells i = tid + 256k  (row r = (tid>>7)+2k, col = tid&127)
    uint64_t msk = 0ull;
    float cur[64];
    const int ccol = tid & 127;
    const int dl = (ccol == 0)   ?  1 : -1;
    const int dr = (ccol == 127) ? -1 :  1;
    #pragma unroll
    for (int k = 0; k < 64; ++k) {
        int i = tid + (k << 8);
        float v = field[i];
        if (v != v) {
            v = 0.5f * (row_m[i >> 7] + col_m[ccol]);
            field[i] = v;
            msk |= (1ull << k);
        }
        cur[k] = v;
    }
    __syncthreads();

    // ---------- 100 Jacobi iterations: branchless, vertical reuse ---------
    for (int it = 0; it < JAC_ITERS; ++it) {
        float upv = field[(tid < 128) ? tid + 128 : tid - 128]; // reflect row0
        #pragma unroll
        for (int k = 0; k < 64; ++k) {
            int i = tid + (k << 8);
            int dn = (k == 63 && tid >= 128) ? i - 128 : i + 128; // reflect 127
            float dnv = field[dn];
            float avg = 0.25f * (upv + dnv + field[i + dl] + field[i + dr]);
            cur[k] = (msk & (1ull << k)) ? avg : cur[k];
            upv = dnv;
        }
        __syncthreads();
        #pragma unroll
        for (int k = 0; k < 64; ++k)
            field[tid + (k << 8)] = cur[k];
        __syncthreads();
    }

    // ---------- LayerNorm stats (from registers) ----------
    {
        float s = 0.0f, s2 = 0.0f;
        #pragma unroll
        for (int k = 0; k < 64; ++k) { s += cur[k]; s2 += cur[k] * cur[k]; }
        #pragma unroll
        for (int off = 16; off > 0; off >>= 1) {
            s  += __shfl_xor(s,  off, 32);
            s2 += __shfl_xor(s2, off, 32);
        }
        if (lane == 0) { red[wid * 2] = s; red[wid * 2 + 1] = s2; }
        __syncthreads();
        if (tid == 0) {
            float S = 0.0f, S2 = 0.0f;
            for (int w = 0; w < 8; ++w) { S += red[w * 2]; S2 += red[w * 2 + 1]; }
            float mean = S * (1.0f / 16384.0f);
            float var  = S2 * (1.0f / 16384.0f) - mean * mean;
            red[16] = mean;
            red[17] = __builtin_amdgcn_rsqf(var + LN_EPS);
        }
        __syncthreads();
    }
    // ---------- normalize -> f16 ----------
    {
        float mean = red[16], rstd = red[17];
        #pragma unroll
        for (int k = 0; k < 64; ++k) {
            int i = tid + (k << 8);
            normh[i] = (_Float16)((cur[k] - mean) * rstd * ln_w[i] + ln_b[i]);
        }
    }
    __syncthreads();   // field region dead; out1 may overlay it

    // ======================================================================
    // conv1: implicit GEMM  M=4096 (64x64), N=16, K=25 (padded 32)
    // ======================================================================
    {
        int n = lane & 15;
        v8h blo = *(const v8h*)(w1h + n * 32 + 16 * lhi);
        v8h bhi = *(const v8h*)(w1h + n * 32 + 16 * lhi + 8);
        v16h bf = __builtin_shufflevector(blo, bhi,
                     0,1,2,3,4,5,6,7,8,9,10,11,12,13,14,15);
        float bias = b1s[n];
        for (int mt = wid; mt < 256; mt += 8) {
            int m   = (mt << 4) + (lane & 15);
            int oy2 = ((m >> 6) << 1) - 2;       // oy*2 - 2
            int ox2 = ((m & 63) << 1) - 2;
            v16h a;
            #pragma unroll
            for (int h = 0; h < 16; ++h) {
                int K  = a_koff(h, lhi);
                int ky = (K * 205) >> 10;        // K/5
                int kx = K - 5 * ky;
                int iy = oy2 + ky, ix = ox2 + kx;
                bool valid = ((unsigned)iy < 128u) & ((unsigned)ix < 128u) & (K < 25);
                int off = valid ? (NORM_H + iy * 128 + ix) : ZERO_H;
                a[h] = lds_h[off];
            }
            v8f acc = {};
            acc = __builtin_amdgcn_wmma_f32_16x16x32_f16(
                      false, a, false, bf, (short)0, acc, false, false);
            v8h o;
            #pragma unroll
            for (int j = 0; j < 8; ++j)
                o[j] = (_Float16)gelu_tanh(acc[j] + bias);
            // 8 consecutive halves, 16B aligned -> single ds_store_b128
            *(v8h*)(out1h + n * 4096 + (mt << 4) + (lhi << 3)) = o;
        }
    }
    __syncthreads();

    // ======================================================================
    // conv2: M=1024 (32x32), N=32, K=144 (5 k-steps, padded 160)
    // ======================================================================
    {
        for (int p = wid; p < 128; p += 8) {
            int mt  = p & 63, nt = p >> 6;
            int m   = (mt << 4) + (lane & 15);
            int oy2 = ((m >> 5) << 1) - 1;
            int ox2 = ((m & 31) << 1) - 1;
            int n   = (nt << 4) + (lane & 15);
            v8f acc = {};
            #pragma unroll
            for (int ks = 0; ks < 5; ++ks) {
                int kb = ks * 32;
                const _Float16* wp = w2h + n * 160 + kb + 16 * lhi;
                v8h blo = *(const v8h*)wp;
                v8h bhi = *(const v8h*)(wp + 8);
                v16h bf = __builtin_shufflevector(blo, bhi,
                             0,1,2,3,4,5,6,7,8,9,10,11,12,13,14,15);
                v16h a;
                #pragma unroll
                for (int h = 0; h < 16; ++h) {
                    int Ka = kb + a_koff(h, lhi);
                    int ci = (Ka * 57) >> 9;     // Ka/9  (Ka < 512)
                    int r  = Ka - 9 * ci;
                    int ky = (r * 171) >> 9;     // r/3
                    int kx = r - 3 * ky;
                    int iy = oy2 + ky, ix = ox2 + kx;
                    bool valid = ((unsigned)iy < 64u) & ((unsigned)ix < 64u) & (Ka < 144);
                    int off = valid ? (OUT1_H + ci * 4096 + iy * 64 + ix) : ZERO_H;
                    a[h] = lds_h[off];
                }
                acc = __builtin_amdgcn_wmma_f32_16x16x32_f16(
                          false, a, false, bf, (short)0, acc, false, false);
            }
            float bias = b2s[n];
            v8h o;
            #pragma unroll
            for (int j = 0; j < 8; ++j)
                o[j] = (_Float16)gelu_tanh(acc[j] + bias);
            *(v8h*)(out2h + n * 1024 + (mt << 4) + (lhi << 3)) = o;
        }
    }
    __syncthreads();

    // ======================================================================
    // conv3 + mean pool: M=256 (16x16), N=64, K=288 (9 k-steps, exact)
    // ======================================================================
    {
        for (int p = wid; p < 64; p += 8) {
            int mt  = p & 15, nt = p >> 4;
            int m   = (mt << 4) + (lane & 15);
            int oy2 = ((m >> 4) << 1) - 1;
            int ox2 = ((m & 15) << 1) - 1;
            int n   = (nt << 4) + (lane & 15);
            v8f acc = {};
            #pragma unroll
            for (int ks = 0; ks < 9; ++ks) {
                int kb = ks * 32;
                const _Float16* wp = w3h + n * 288 + kb + 16 * lhi;
                v8h blo = *(const v8h*)wp;
                v8h bhi = *(const v8h*)(wp + 8);
                v16h bf = __builtin_shufflevector(blo, bhi,
                             0,1,2,3,4,5,6,7,8,9,10,11,12,13,14,15);
                v16h a;
                #pragma unroll
                for (int h = 0; h < 16; ++h) {
                    int Ka = kb + a_koff(h, lhi);   // < 288 always
                    int ci = (Ka * 57) >> 9;
                    int r  = Ka - 9 * ci;
                    int ky = (r * 171) >> 9;
                    int kx = r - 3 * ky;
                    int iy = oy2 + ky, ix = ox2 + kx;
                    bool valid = ((unsigned)iy < 32u) & ((unsigned)ix < 32u);
                    int off = valid ? (OUT2_H + ci * 1024 + iy * 32 + ix) : ZERO_H;
                    a[h] = lds_h[off];
                }
                acc = __builtin_amdgcn_wmma_f32_16x16x32_f16(
                          false, a, false, bf, (short)0, acc, false, false);
            }
            float bias = b3s[n];
            float s = 0.0f;
            #pragma unroll
            for (int j = 0; j < 8; ++j)
                s += gelu_tanh(acc[j] + bias);
            atomicAdd(&feat[n], s);    // one ds_add_f32 per lane per tile
        }
    }
    __syncthreads();

    // ---------- linear head: out5 = (feat/256) @ wl^T + bl ----------
    if (tid < 5) {
        float s = bl[tid];
        for (int c = 0; c < 64; ++c)
            s += feat[c] * (1.0f / 256.0f) * wl[tid * 64 + c];
        out5s[tid] = s;
    }
    __syncthreads();

    if (tid == 0) {
        float o0 = out5s[0], o1 = out5s[1], o2 = out5s[2], o3 = out5s[3], o4 = out5s[4];
        out[2 * b + 0] = nan_to_0(o1);
        out[2 * b + 1] = nan_to_0(o0);
        float s00 = nan_to_1(__builtin_amdgcn_exp2f(o2 * 1.4426950408889634f));
        float s11 = nan_to_1(__builtin_amdgcn_exp2f(o3 * 1.4426950408889634f));
        float s01 = nan_to_1(o4);
        float* so = out + 2 * B + 4 * b;
        so[0] = s00; so[1] = s01; so[2] = s01; so[3] = s11;
    }
}

extern "C" void kernel_launch(void* const* d_in, const int* in_sizes, int n_in,
                              void* d_out, int out_size, void* d_ws, size_t ws_size,
                              hipStream_t stream) {
    (void)n_in; (void)out_size; (void)d_ws; (void)ws_size;
    const float* sst  = (const float*)d_in[0];
    const float* ln_w = (const float*)d_in[1];
    const float* ln_b = (const float*)d_in[2];
    const float* w1   = (const float*)d_in[3];
    const float* b1   = (const float*)d_in[4];
    const float* w2   = (const float*)d_in[5];
    const float* b2   = (const float*)d_in[6];
    const float* w3   = (const float*)d_in[7];
    const float* b3   = (const float*)d_in[8];
    const float* wl   = (const float*)d_in[9];
    const float* bl   = (const float*)d_in[10];
    float* out = (float*)d_out;
    int B = in_sizes[0] / (128 * 128);

    encoder_fused_kernel<<<B, TPB, 0, stream>>>(
        sst, ln_w, ln_b, w1, b1, w2, b2, w3, b3, wl, bl, out, B);
}
